// Dilatedconv_WindowAttention_77910706749739
// MI455X (gfx1250) — matrix operand order repaired and checked by reading, not compiled, not run on hardware
//
#include <hip/hip_runtime.h>
#include <hip/hip_bf16.h>
#include <cstdint>

typedef __attribute__((ext_vector_type(16))) _Float16 v16h;
typedef __attribute__((ext_vector_type(8)))  _Float16 v8h;
typedef __attribute__((ext_vector_type(8)))  float    v8f;

#define BATCH 4096
#define SEQ   49
#define CH    256
#define HEADS 8
#define HD    32
#define MKV   21
#define QSCALE 0.17677669529663687f   // 32^-0.5

// LDS layout (bytes):
//  xh    f16 [64][256]      @ 0       (32768)
//  qh    f16 [64][256]      @ 32768   (32768)  q first, attention-out later
//  dpad  f16 [4][16][256]   @ 65536   (32768)  per-branch zero-padded conv out
//  kf    f16 [32][256]      @ 98304   (16384)  k, row = kv index, col = channel
//  vt    f16 [256][32]      @ 114688  (16384)  v transposed: [channel][kv index]
//  Pbuf  f16 [8][64][32]    @ 131072  (32768)  softmax probabilities
//  Sbuf  f32 [8][64][32]    @ 163840  (65536)  raw scores
#define LDS_BYTES 229376

__device__ __forceinline__ v16h cat8(v8h lo, v8h hi) {
  return __builtin_shufflevector(lo, hi, 0,1,2,3,4,5,6,7,8,9,10,11,12,13,14,15);
}

// A fragment (16x32 f16): lane l -> row m0+(l&15); K = 8*(l>>4)+e (e<8), 16+8*(l>>4)+(e-8)
__device__ __forceinline__ v16h loadA16(const _Float16* base, int ld, int m0, int lane) {
  const _Float16* p = base + (m0 + (lane & 15)) * ld + 8 * (lane >> 4);
  v8h lo = *(const v8h*)p;
  v8h hi = *(const v8h*)(p + 16);
  return cat8(lo, hi);
}

// B fragment (32x16 f16, K contiguous within a column image): lane l -> col n0+(l&15); K = 8*(l>>4)+e (e<8), 16+8*(l>>4)+(e-8)
__device__ __forceinline__ v16h loadB16(const _Float16* base, int ld, int n0, int lane) {
  const _Float16* p = base + (n0 + (lane & 15)) * ld + 8 * (lane >> 4);
  v8h lo = *(const v8h*)p;
  v8h hi = *(const v8h*)(p + 16);
  return cat8(lo, hi);
}

__device__ __forceinline__ v8f wmma32(v16h a, v16h b, v8f c) {
  return __builtin_amdgcn_wmma_f32_16x16x32_f16(false, a, false, b, (short)0, c, false, false);
}

__global__ void cvt_f32_f16(const float* __restrict__ s, _Float16* __restrict__ d, int n) {
  int i = blockIdx.x * blockDim.x + threadIdx.x;
  if (i < n) d[i] = (_Float16)s[i];
}

__global__ __launch_bounds__(256, 1)
void fused_dilated_attn(
    const float* __restrict__ x,
    const float* __restrict__ bq,
    const float* __restrict__ wd0, const float* __restrict__ bd0,
    const float* __restrict__ wd1, const float* __restrict__ bd1,
    const float* __restrict__ wd2, const float* __restrict__ bd2,
    const float* __restrict__ wd3, const float* __restrict__ bd3,
    const float* __restrict__ bp0, const float* __restrict__ bp1,
    const float* __restrict__ bp2, const float* __restrict__ bp3,
    const float* __restrict__ bproj,
    const _Float16* __restrict__ wq16,
    const _Float16* __restrict__ wp16,      // 4 consecutive (512*256) blocks
    const _Float16* __restrict__ wproj16,
    float* __restrict__ out)
{
  extern __shared__ float4 smem4[];
  char* smem = (char*)smem4;
  _Float16* xh   = (_Float16*)smem;            // 64*256
  _Float16* qh   = xh   + 64 * 256;            // 64*256
  _Float16* dpad = qh   + 64 * 256;            // 4*16*256
  _Float16* kf   = dpad + 4 * 16 * 256;        // 32*256
  _Float16* vt   = kf   + 32 * 256;            // 256*32
  _Float16* Pbuf = vt   + 256 * 32;            // 8*64*32
  float*    Sbuf = (float*)(Pbuf + 8 * 64 * 32); // 8*64*32

  const int t    = threadIdx.x;
  const int lane = t & 31;
  const int wv   = t >> 5;        // 0..7 : wave id == head id in attention
  const int half = lane >> 4;
  const int b    = blockIdx.x;

  // ---- warm GL2 with the f16 weight images (global_prefetch path) ----
  __builtin_prefetch(wq16 + (size_t)t * 256, 0, 1);
  __builtin_prefetch(wp16 + (size_t)t * 2048, 0, 1);
  __builtin_prefetch(wproj16 + (size_t)t * 256, 0, 1);

  // ---- zero only the LDS regions whose padding matters for WMMA tiles ----
  {
    uint32_t* w = (uint32_t*)smem;
    // xh pad rows 49..63: bytes [25088, 32768) -> words [6272, 8192)
    for (int i = 6272 + t; i < 8192; i += 256) w[i] = 0u;
    // dpad + kf + vt + Pbuf: bytes [65536, 163840) -> words [16384, 40960)
    for (int i = 16384 + t; i < 40960; i += 256) w[i] = 0u;
  }
  __syncthreads();

  // ---- load x_b -> f16 LDS ----
  const float* xb = x + (size_t)b * (SEQ * CH);
  for (int i = t; i < SEQ * CH; i += 256) xh[i] = (_Float16)xb[i];
  __syncthreads();

  // ---- q = (x @ wq^T + bq) * scale ; M=64(pad), N=256, K=256 ----
  // B strip (8 K-tiles) hoisted into registers per nt; M-tiles swept in pairs
  // with independent accumulators (back-to-back WMMA, no RAW hazard).
  for (int nt = wv; nt < 16; nt += 8) {
    v16h bs[8];
#pragma unroll
    for (int k = 0; k < 8; ++k) bs[k] = loadB16(wq16 + k * 32, CH, nt * 16, lane);
    int o = nt * 16 + (lane & 15);
    float bias = bq[o];
    for (int mt = 0; mt < 4; mt += 2) {
      v8f acc0 = {}, acc1 = {};
#pragma unroll
      for (int k = 0; k < 8; ++k) {
        v16h a0 = loadA16(xh + k * 32, CH, mt * 16, lane);
        v16h a1 = loadA16(xh + k * 32, CH, mt * 16 + 16, lane);
        acc0 = wmma32(a0, bs[k], acc0);
        acc1 = wmma32(a1, bs[k], acc1);
      }
#pragma unroll
      for (int r = 0; r < 8; ++r) {
        int row0 = mt * 16 + r + 8 * half;
        qh[row0 * CH + o]        = (_Float16)((acc0[r] + bias) * QSCALE);
        qh[(row0 + 16) * CH + o] = (_Float16)((acc1[r] + bias) * QSCALE);
      }
    }
  }

  // ---- depthwise 3x3 stride-2 dilated convs (21 positions x 256 ch) ----
  for (int f = t; f < MKV * CH; f += 256) {
    int pg = f >> 8;          // global kv position 0..20
    int c  = f & 255;
    int p, ow, dil, pad, jb;
    const float* wd; const float* bd;
    if (pg < 9)       { jb = 0; p = pg;      ow = 3; dil = 1; pad = 0; wd = wd0; bd = bd0; }
    else if (pg < 13) { jb = 1; p = pg - 9;  ow = 2; dil = 2; pad = 0; wd = wd1; bd = bd1; }
    else if (pg < 17) { jb = 2; p = pg - 13; ow = 2; dil = 3; pad = 1; wd = wd2; bd = bd2; }
    else              { jb = 3; p = pg - 17; ow = 2; dil = 4; pad = 2; wd = wd3; bd = bd3; }
    int oy = p / ow, ox = p % ow;
    float acc = bd[c];
    for (int ky = 0; ky < 3; ++ky) {
      int iy = oy * 2 - pad + ky * dil;
      if (iy < 0 || iy >= 7) continue;
      for (int kx = 0; kx < 3; ++kx) {
        int ix = ox * 2 - pad + kx * dil;
        if (ix < 0 || ix >= 7) continue;
        acc += (float)xh[(iy * 7 + ix) * CH + c] * wd[c * 9 + ky * 3 + kx];
      }
    }
    dpad[(jb * 16 + p) * CH + c] = (_Float16)acc;
  }
  __syncthreads();

  // ---- pointwise: kv_j = d_j @ wp_j^T + bp_j, scatter into kf / vt ----
  // reference channel/spatial shuffle: u = o*hw + s ; c' = u/(2hw); t = u%(2hw);
  // g = t&1 (0->k, 1->v); s' = t>>1
  for (int job = wv; job < 128; job += 8) {
    int jb = job >> 5;          // branch 0..3
    int nt = job & 31;          // output tile over 512
    const _Float16* wpj = wp16 + jb * (512 * 256);
    const float* bpj = jb == 0 ? bp0 : jb == 1 ? bp1 : jb == 2 ? bp2 : bp3;
    int hwj = jb == 0 ? 9 : 4;
    int off = jb == 0 ? 0 : 9 + (jb - 1) * 4;
    v8f acc = {};
#pragma unroll
    for (int k = 0; k < 8; ++k) {
      v16h a  = loadA16(dpad + jb * 16 * CH + k * 32, CH, 0, lane);
      v16h bm = loadB16(wpj + k * 32, CH, nt * 16, lane);
      acc = wmma32(a, bm, acc);
    }
    int o = nt * 16 + (lane & 15);
    float bias = bpj[o];
    int twoHw = 2 * hwj;
#pragma unroll
    for (int r = 0; r < 8; ++r) {
      int m = r + 8 * half;
      if (m >= hwj) continue;
      float val = acc[r] + bias;
      int u  = o * hwj + m;
      int cp = u / twoHw;
      int tt = u % twoHw;
      int sp = tt >> 1;
      if ((tt & 1) == 0) kf[(off + sp) * CH + cp] = (_Float16)val;
      else               vt[cp * 32 + (off + sp)] = (_Float16)val;
    }
  }
  __syncthreads();

  // ---- S = q_h @ k_h^T per head (wave == head); K = 32, one WMMA step ----
  {
    float* Sh = Sbuf + wv * (64 * 32);
    const _Float16* qbase = qh + wv * HD;
    const _Float16* kbase = kf + wv * HD;
    v16h b0 = loadB16(kbase, CH, 0, lane);
    v16h b1 = loadB16(kbase, CH, 16, lane);
    for (int mt = 0; mt < 4; ++mt) {
      v16h a = loadA16(qbase, CH, mt * 16, lane);
      v8f acc0 = {}, acc1 = {};
      acc0 = wmma32(a, b0, acc0);
      acc1 = wmma32(a, b1, acc1);
      int col = lane & 15;
#pragma unroll
      for (int r = 0; r < 8; ++r) {
        int row = mt * 16 + r + 8 * half;
        Sh[row * 32 + col]      = acc0[r];
        Sh[row * 32 + col + 16] = acc1[r];
      }
    }
  }
  __syncthreads();

  // ---- softmax over 21 kv positions, write P (f16, zero-padded to 32) ----
  for (int idx = t; idx < HEADS * SEQ; idx += 256) {
    int hh = idx / SEQ, row = idx % SEQ;
    const float* sr = Sbuf + hh * 2048 + row * 32;
    _Float16*    pr = Pbuf + hh * 2048 + row * 32;
    float mx = -1e30f;
    for (int m = 0; m < MKV; ++m) mx = fmaxf(mx, sr[m]);
    float sum = 0.f;
    for (int m = 0; m < MKV; ++m) sum += __expf(sr[m] - mx);
    float inv = 1.0f / sum;
    for (int m = 0; m < MKV; ++m) pr[m] = (_Float16)(__expf(sr[m] - mx) * inv);
    for (int m = MKV; m < 32; ++m) pr[m] = (_Float16)0.f;
  }
  __syncthreads();

  // ---- out_h = P @ v ; K = 32 (kv index, padded with zeros) ----
  {
    const _Float16* pb = Pbuf + wv * 2048;
    const _Float16* vb = vt + wv * HD * 32;   // vt[channel][kv]
    v16h b0 = loadB16(vb, 32, 0, lane);
    v16h b1 = loadB16(vb, 32, 16, lane);
    for (int mt = 0; mt < 4; ++mt) {
      v16h a = loadA16(pb, 32, mt * 16, lane);
      v8f acc0 = {}, acc1 = {};
      acc0 = wmma32(a, b0, acc0);
      acc1 = wmma32(a, b1, acc1);
      int col = lane & 15;                    // head-dim index
#pragma unroll
      for (int r = 0; r < 8; ++r) {
        int row = mt * 16 + r + 8 * half;
        qh[row * CH + wv * HD + col]      = (_Float16)acc0[r];  // qh reused as out
        qh[row * CH + wv * HD + col + 16] = (_Float16)acc1[r];
      }
    }
  }
  __syncthreads();

  // ---- y = out @ wproj^T + bproj ; store rows < 49 as f32 ----
  for (int nt = wv; nt < 16; nt += 8) {
    v16h bs[8];
#pragma unroll
    for (int k = 0; k < 8; ++k) bs[k] = loadB16(wproj16 + k * 32, CH, nt * 16, lane);
    int o = nt * 16 + (lane & 15);
    float bias = bproj[o];
    for (int mt = 0; mt < 4; mt += 2) {
      v8f acc0 = {}, acc1 = {};
#pragma unroll
      for (int k = 0; k < 8; ++k) {
        v16h a0 = loadA16(qh + k * 32, CH, mt * 16, lane);
        v16h a1 = loadA16(qh + k * 32, CH, mt * 16 + 16, lane);
        acc0 = wmma32(a0, bs[k], acc0);
        acc1 = wmma32(a1, bs[k], acc1);
      }
#pragma unroll
      for (int r = 0; r < 8; ++r) {
        int row0 = mt * 16 + r + 8 * half;
        int row1 = row0 + 16;
        if (row0 < SEQ) out[((size_t)b * SEQ + row0) * CH + o] = acc0[r] + bias;
        if (row1 < SEQ) out[((size_t)b * SEQ + row1) * CH + o] = acc1[r] + bias;
      }
    }
  }
}

extern "C" void kernel_launch(void* const* d_in, const int* in_sizes, int n_in,
                              void* d_out, int out_size, void* d_ws, size_t ws_size,
                              hipStream_t stream) {
  (void)in_sizes; (void)n_in; (void)out_size; (void)ws_size;
  const float* x  = (const float*)d_in[0];
  const float* wq = (const float*)d_in[1];
  const float* bq = (const float*)d_in[2];
  const float *wd[4], *bd[4], *wp[4], *bp[4];
  for (int j = 0; j < 4; ++j) {
    wd[j] = (const float*)d_in[3 + 4 * j];
    bd[j] = (const float*)d_in[4 + 4 * j];
    wp[j] = (const float*)d_in[5 + 4 * j];
    bp[j] = (const float*)d_in[6 + 4 * j];
  }
  const float* wproj = (const float*)d_in[19];
  const float* bproj = (const float*)d_in[20];

  // f16 weight copies in workspace (stay L2-resident across all 4096 WGs)
  _Float16* wq16    = (_Float16*)d_ws;              // 65536
  _Float16* wp16    = wq16 + 65536;                 // 4 * 131072
  _Float16* wproj16 = wp16 + 4 * 131072;            // 65536

  cvt_f32_f16<<<(65536 + 255) / 256, 256, 0, stream>>>(wq, wq16, 65536);
  for (int j = 0; j < 4; ++j)
    cvt_f32_f16<<<(131072 + 255) / 256, 256, 0, stream>>>(wp[j], wp16 + j * 131072, 131072);
  cvt_f32_f16<<<(65536 + 255) / 256, 256, 0, stream>>>(wproj, wproj16, 65536);

  hipFuncSetAttribute((const void*)fused_dilated_attn,
                      hipFuncAttributeMaxDynamicSharedMemorySize, LDS_BYTES);

  fused_dilated_attn<<<BATCH, 256, LDS_BYTES, stream>>>(
      x, bq,
      wd[0], bd[0], wd[1], bd[1], wd[2], bd[2], wd[3], bd[3],
      bp[0], bp[1], bp[2], bp[3], bproj,
      wq16, wp16, wproj16, (float*)d_out);
}
